// ViTSelfAttention_26585847562913
// MI455X (gfx1250) — matrix operand order repaired and checked by reading, not compiled, not run on hardware
//
#include <hip/hip_runtime.h>
#include <hip/hip_bf16.h>
#include <math.h>

// ---------------------------------------------------------------------------
// ViT self-attention for MI455X (gfx1250, wave32, WMMA).
// All GEMM work runs on v_wmma_f32_16x16x32_bf16 (f32 accumulate).
// Attention is flash-style: scores never touch HBM; softmax in f32.
// GEMM uses 4x2 WMMA register blocking per wave (64x32 output tile) so each
// fragment load feeds 2-4 WMMAs (~21 FLOP/byte from cache) instead of 2.
// ---------------------------------------------------------------------------

#define D_MODEL  768
#define N_HEADS  12
#define HEAD_DIM 64
#define BATCH    16
#define SEQ      1024
#define MROWS    (BATCH * SEQ)   // 16384

typedef __attribute__((ext_vector_type(16))) __bf16 v16bf;
typedef __attribute__((ext_vector_type(8)))  float  v8f;

union FragU { uint4 q[2]; v16bf v; };

// f32 -> bf16, round-to-nearest-even
static __device__ __forceinline__ unsigned short f2bf(float f) {
  unsigned u = __float_as_uint(f);
  u += 0x7FFFu + ((u >> 16) & 1u);
  return (unsigned short)(u >> 16);
}

// A-matrix fragment, 16x32 bf16 (ISA 7.12.2):
// lanes 0-15: row M=lane, elems = K {k0+0..7, k0+16..23}
// lanes 16-31: row M=lane-16, elems = K {k0+8..15, k0+24..31}
static __device__ __forceinline__ v16bf load_a_frag(const unsigned short* base,
                                                    int ld, int lane, int k0) {
  const int r  = lane & 15;
  const int kb = k0 + ((lane & 16) ? 8 : 0);
  const unsigned short* p = base + (size_t)r * ld + kb;
  FragU u;
  u.q[0] = *(const uint4*)(p);        // K = kb .. kb+7
  u.q[1] = *(const uint4*)(p + 16);   // K = kb+16 .. kb+23
  return u.v;
}

// B-matrix fragment, 32x16 bf16 (same K order as A fragment):
// lanes 0-15: col N=lane, elems = K {k0+0..7, k0+16..23} ; lanes 16-31: K {k0+8..15, k0+24..31}
static __device__ __forceinline__ v16bf load_b_frag(const unsigned short* base,
                                                    int ld, int lane, int k0) {
  const int c  = lane & 15;
  const int kb = k0 + ((lane & 16) ? 8 : 0);
  const unsigned short* p = base + (size_t)c * ld + kb;
  FragU u;
  u.q[0] = *(const uint4*)(p);        // K = kb .. kb+7
  u.q[1] = *(const uint4*)(p + 16);   // K = kb+16 .. kb+23
  return u.v;
}

static __device__ __forceinline__ v8f wmma_bf16(v16bf a, v16bf b, v8f c) {
  return __builtin_amdgcn_wmma_f32_16x16x32_bf16(
      /*neg_a=*/false, a, /*neg_b=*/false, b,
      /*c_mod=*/(short)0, c, /*reuse_a=*/false, /*reuse_b=*/false);
}

// ---------------------------------------------------------------------------
// f32 -> bf16 conversion (x and the four weight matrices)
// ---------------------------------------------------------------------------
__global__ void cvt_f32_bf16(const float* __restrict__ src,
                             unsigned short* __restrict__ dst, int n) {
  for (int i = blockIdx.x * blockDim.x + threadIdx.x; i < n;
       i += gridDim.x * blockDim.x)
    dst[i] = f2bf(src[i]);
}

// ---------------------------------------------------------------------------
// C[M,N] = A[M,K] * W[N,K]^T + bias.  256 threads = 8 waves.
// Wave tile: 64x32 = 4x2 WMMA tiles (8 WMMA per K-step, 6 fragment loads).
// Waves arranged 2(M) x 4(N) -> block tile 128x128.
// mode 0: store bf16 head-split [B,H,S,Dh] with scale (QKV path)
// mode 1: store f32 [M, D_MODEL] (output projection)
// ---------------------------------------------------------------------------
__global__ __launch_bounds__(256) void gemm_bf16_nt(
    const unsigned short* __restrict__ A,
    const unsigned short* __restrict__ W,
    const float* __restrict__ bias,
    unsigned short* __restrict__ outHS,
    float* __restrict__ outF,
    float scale, int K, int mode) {
  const int lane = threadIdx.x & 31;
  const int wave = threadIdx.x >> 5;
  const int wm = wave & 1;                      // 2 waves across M
  const int wn = wave >> 1;                     // 4 waves across N
  const int m0 = blockIdx.x * 128 + wm * 64;
  const int n0 = blockIdx.y * 128 + wn * 32;

  v8f acc[4][2] = {};
  for (int k = 0; k < K; k += 32) {
    v16bf a[4], b[2];
#pragma unroll
    for (int mi = 0; mi < 4; ++mi)
      a[mi] = load_a_frag(A + (size_t)(m0 + mi * 16) * K, K, lane, k);
#pragma unroll
    for (int ni = 0; ni < 2; ++ni)
      b[ni] = load_b_frag(W + (size_t)(n0 + ni * 16) * K, K, lane, k);
#pragma unroll
    for (int mi = 0; mi < 4; ++mi)
#pragma unroll
      for (int ni = 0; ni < 2; ++ni)
        acc[mi][ni] = wmma_bf16(a[mi], b[ni], acc[mi][ni]);
  }

  const int hi = (lane >> 4) & 1;   // lanes 16-31 hold rows M+8 in C layout
#pragma unroll
  for (int ni = 0; ni < 2; ++ni) {
    const int n = n0 + ni * 16 + (lane & 15);
    const float bn = bias ? bias[n] : 0.f;
#pragma unroll
    for (int mi = 0; mi < 4; ++mi) {
#pragma unroll
      for (int r = 0; r < 8; ++r) {
        const int m = m0 + mi * 16 + r + hi * 8;
        const float v = acc[mi][ni][r] + bn;
        if (mode == 0) {
          const int b = m >> 10, s = m & 1023;
          const int h = n >> 6,  d = n & 63;
          outHS[(((size_t)(b * N_HEADS + h) << 10) + s) * HEAD_DIM + d] =
              f2bf(v * scale);
        } else {
          outF[(size_t)m * D_MODEL + n] = v;
        }
      }
    }
  }
}

// ---------------------------------------------------------------------------
// Flash attention: one block = (head bh, 64 query rows). 128 threads = 4
// waves; each wave owns 16 query rows. Online softmax kept entirely in the
// WMMA C-layout (rows = VGPR idx + lane half, cols = 16 lanes).
// Q is pre-scaled by 1/sqrt(Dh).
// ---------------------------------------------------------------------------
__global__ __launch_bounds__(128) void flash_attn(
    const unsigned short* __restrict__ Q,   // [B*H, S, 64] bf16
    const unsigned short* __restrict__ Kk,  // [B*H, S, 64] bf16
    const unsigned short* __restrict__ V,   // [B*H, S, 64] bf16
    unsigned short* __restrict__ ctx) {     // [B, S, 768] bf16
  __shared__ __align__(16) unsigned short lds_k [64 * 64];      // K block row-major
  __shared__ __align__(16) unsigned short lds_vt[64 * 64];      // V block transposed [d][k]
  __shared__ __align__(16) unsigned short lds_p [4 * 16 * 64];  // per-wave P (16x64)

  const int bh   = blockIdx.x;        // 0..191
  const int tid  = threadIdx.x;
  const int lane = tid & 31;
  const int wave = tid >> 5;
  const int hi   = (lane >> 4) & 1;

  const unsigned short* Qh = Q  + (size_t)bh * SEQ * HEAD_DIM;
  const unsigned short* Kh = Kk + (size_t)bh * SEQ * HEAD_DIM;
  const unsigned short* Vh = V  + (size_t)bh * SEQ * HEAD_DIM;

  const int qrow0 = blockIdx.y * 64 + wave * 16;

  // Q fragments (16x64 = two 16x32 A fragments), loaded once
  const v16bf qa0 = load_a_frag(Qh + (size_t)qrow0 * HEAD_DIM, HEAD_DIM, lane, 0);
  const v16bf qa1 = load_a_frag(Qh + (size_t)qrow0 * HEAD_DIM, HEAD_DIM, lane, 32);

  v8f oacc[4] = {};
  float mrun[8], lrun[8];
#pragma unroll
  for (int r = 0; r < 8; ++r) { mrun[r] = -1e30f; lrun[r] = 0.f; }

  unsigned short* pbuf = lds_p + wave * 16 * 64;

  for (int kb = 0; kb < SEQ / 64; ++kb) {
    // ---- stage K (row-major) and V (transposed) into LDS -----------------
    {
      const uint4* ksrc = (const uint4*)(Kh + (size_t)kb * 64 * HEAD_DIM);
      uint4* kdst = (uint4*)lds_k;
#pragma unroll
      for (int i = 0; i < 4; ++i) kdst[tid + i * 128] = ksrc[tid + i * 128];
      const unsigned short* vsrc = Vh + (size_t)kb * 64 * HEAD_DIM;
      for (int i = tid; i < 64 * 64; i += 128) {
        const int kk = i >> 6, dd = i & 63;
        lds_vt[dd * 64 + kk] = vsrc[i];
      }
    }
    if (kb + 1 < SEQ / 64) {   // global_prefetch_b8 next K/V block
      __builtin_prefetch(Kh + (size_t)(kb + 1) * 64 * HEAD_DIM + tid * 32, 0, 1);
      __builtin_prefetch(Vh + (size_t)(kb + 1) * 64 * HEAD_DIM + tid * 32, 0, 1);
    }
    __syncthreads();

    // ---- S(16x64) = Q * K^T (Q already carries 1/sqrt(Dh)) ---------------
    v8f sacc[4] = {};
#pragma unroll
    for (int nt = 0; nt < 4; ++nt) {
      v16bf b0 = load_b_frag(lds_k + nt * 16 * 64, 64, lane, 0);
      v16bf b1 = load_b_frag(lds_k + nt * 16 * 64, 64, lane, 32);
      sacc[nt] = wmma_bf16(qa0, b0, sacc[nt]);
      sacc[nt] = wmma_bf16(qa1, b1, sacc[nt]);
    }

    // ---- online softmax ---------------------------------------------------
    // row r lives in VGPR r; lanes 0-15 carry rows 0-7, lanes 16-31 rows 8-15.
    // Row reductions: 4 tiles in-lane, then shfl_xor 1/2/4/8 within each half.
    float alpha[8];
#pragma unroll
    for (int r = 0; r < 8; ++r) {
      float mx = fmaxf(fmaxf(sacc[0][r], sacc[1][r]),
                       fmaxf(sacc[2][r], sacc[3][r]));
#pragma unroll
      for (int msk = 1; msk <= 8; msk <<= 1)
        mx = fmaxf(mx, __shfl_xor(mx, msk, 32));
      const float mnew = fmaxf(mrun[r], mx);
      alpha[r] = __expf(mrun[r] - mnew);
      mrun[r]  = mnew;
    }

    float lsum[8];
#pragma unroll
    for (int r = 0; r < 8; ++r) lsum[r] = 0.f;
#pragma unroll
    for (int nt = 0; nt < 4; ++nt) {
      const int col = nt * 16 + (lane & 15);
#pragma unroll
      for (int r = 0; r < 8; ++r) {
        const float p = __expf(sacc[nt][r] - mrun[r]);
        lsum[r] += p;
        pbuf[(r + hi * 8) * 64 + col] = f2bf(p);   // C-layout -> LDS
      }
    }
#pragma unroll
    for (int r = 0; r < 8; ++r) {
      float s = lsum[r];
#pragma unroll
      for (int msk = 1; msk <= 8; msk <<= 1) s += __shfl_xor(s, msk, 32);
      lrun[r] = lrun[r] * alpha[r] + s;
    }

    // ---- O = O*alpha + P*V  (P re-read from LDS in A layout) -------------
    const v16bf pa0 = load_a_frag(pbuf, 64, lane, 0);
    const v16bf pa1 = load_a_frag(pbuf, 64, lane, 32);
#pragma unroll
    for (int nt = 0; nt < 4; ++nt) {
#pragma unroll
      for (int r = 0; r < 8; ++r) oacc[nt][r] *= alpha[r];
      v16bf vb0 = load_b_frag(lds_vt + nt * 16 * 64, 64, lane, 0);
      v16bf vb1 = load_b_frag(lds_vt + nt * 16 * 64, 64, lane, 32);
      oacc[nt] = wmma_bf16(pa0, vb0, oacc[nt]);
      oacc[nt] = wmma_bf16(pa1, vb1, oacc[nt]);
    }
    __syncthreads();
  }

  // ---- normalize and store ctx (merged-head [B,S,768] bf16) --------------
  const int b = bh / N_HEADS, h = bh % N_HEADS;
#pragma unroll
  for (int nt = 0; nt < 4; ++nt) {
    const int d = nt * 16 + (lane & 15);
#pragma unroll
    for (int r = 0; r < 8; ++r) {
      const int s = qrow0 + r + hi * 8;
      const float o = oacc[nt][r] / lrun[r];
      ctx[((size_t)b * SEQ + s) * D_MODEL + h * HEAD_DIM + d] = f2bf(o);
    }
  }
}

// ---------------------------------------------------------------------------
extern "C" void kernel_launch(void* const* d_in, const int* in_sizes, int n_in,
                              void* d_out, int out_size, void* d_ws, size_t ws_size,
                              hipStream_t stream) {
  const float* x  = (const float*)d_in[0];
  const float* Wq = (const float*)d_in[1];
  const float* bq = (const float*)d_in[2];
  const float* Wk = (const float*)d_in[3];
  const float* bk = (const float*)d_in[4];
  const float* Wv = (const float*)d_in[5];
  const float* bv = (const float*)d_in[6];
  const float* Wc = (const float*)d_in[7];
  const float* bc = (const float*)d_in[8];
  float* out = (float*)d_out;

  const size_t MD = (size_t)MROWS * D_MODEL;     // 12,582,912
  const size_t WW = (size_t)D_MODEL * D_MODEL;   // 589,824

  unsigned short* ws  = (unsigned short*)d_ws;
  unsigned short* xb  = ws;            // [M, D] bf16
  unsigned short* wqb = xb  + MD;      // [D, D] bf16
  unsigned short* wkb = wqb + WW;
  unsigned short* wvb = wkb + WW;
  unsigned short* wcb = wvb + WW;
  unsigned short* Qb  = wcb + WW;      // [B,H,S,Dh] bf16 (pre-scaled)
  unsigned short* Kb  = Qb  + MD;
  unsigned short* Vb  = Kb  + MD;
  unsigned short* Cb  = Vb  + MD;      // ctx [B,S,D] bf16

  cvt_f32_bf16<<<2048, 256, 0, stream>>>(x,  xb,  (int)MD);
  cvt_f32_bf16<<<576,  256, 0, stream>>>(Wq, wqb, (int)WW);
  cvt_f32_bf16<<<576,  256, 0, stream>>>(Wk, wkb, (int)WW);
  cvt_f32_bf16<<<576,  256, 0, stream>>>(Wv, wvb, (int)WW);
  cvt_f32_bf16<<<576,  256, 0, stream>>>(Wc, wcb, (int)WW);

  const dim3 gg(MROWS / 128, D_MODEL / 128);  // (128, 6), block tile 128x128
  const float qscale = 0.125f;                // 1/sqrt(HEAD_DIM)
  gemm_bf16_nt<<<gg, 256, 0, stream>>>(xb, wqb, bq, Qb, nullptr, qscale, D_MODEL, 0);
  gemm_bf16_nt<<<gg, 256, 0, stream>>>(xb, wkb, bk, Kb, nullptr, 1.0f,   D_MODEL, 0);
  gemm_bf16_nt<<<gg, 256, 0, stream>>>(xb, wvb, bv, Vb, nullptr, 1.0f,   D_MODEL, 0);

  flash_attn<<<dim3(BATCH * N_HEADS, SEQ / 64), 128, 0, stream>>>(Qb, Kb, Vb, Cb);

  gemm_bf16_nt<<<gg, 256, 0, stream>>>(Cb, wcb, bc, nullptr, out, 1.0f, D_MODEL, 1);
}
